// NRIMLPEncoder_9397388444175
// MI455X (gfx1250) — compile-verified, matrix operand order checked by reading, not run on hardware
//
#include <hip/hip_runtime.h>
#include <hip/hip_bf16.h>

#define HD     128
#define NNODE  64
#define NEDGE  4032
#define NBATCH 64

typedef __attribute__((ext_vector_type(16))) __bf16 v16bf;
typedef __attribute__((ext_vector_type(4)))  __bf16 v4bf;
typedef __attribute__((ext_vector_type(8)))  float  v8f;

static __device__ inline __bf16 f2bf(float f) {
  union { float f; unsigned u; } x; x.f = f;
  unsigned u = x.u;
  unsigned r = (u + 0x7FFFu + ((u >> 16) & 1u)) >> 16;   // round-to-nearest-even
  union { unsigned short s; __bf16 b; } y; y.s = (unsigned short)r;
  return y.b;
}

enum { MODE_PLAIN = 0, MODE_N2E = 1, MODE_OUT = 2 };

// C = ELU(A @ W + bias). A rows gathered per MODE:
//  PLAIN: A[r*K + k]
//  N2E:   concat(node[recv[e]], node[send[e]])                (K=256)
//  OUT:   concat(node[recv[e]], node[send[e]], skip[r])       (K=384)
//
// LDS tiles are stored directly in WMMA fragment layout:
//   frag slot (grp, lane, i) at lXfrag[grp*32 + lane][i], 32B/lane, so the
//   per-wave operand build is a single 32-byte contiguous LDS load.
// Fragment element i of lane (half*16+lrow) holds k = (i/8)*16 + half*8 + i%8,
// row/col = 16*grp + lrow.  A k-quad (k%4==0) -> same lane, i..i+3.
template<int MODE>
__global__ __launch_bounds__(256) void gemm_bias_elu(
    const float* __restrict__ A, const float* __restrict__ nodeSrc,
    const float* __restrict__ W, const float* __restrict__ bias,
    const int* __restrict__ ridx, const int* __restrict__ sidx,
    float* __restrict__ out, int M, int K)
{
  __shared__ v16bf lAfrag[256];   // 8 wave-tiles x 32 lanes  (8 KB)
  __shared__ v16bf lBfrag[256];   // 8 n-tiles   x 32 lanes  (8 KB)
  __bf16* lAe = (__bf16*)lAfrag;
  __bf16* lBe = (__bf16*)lBfrag;

  const int  tid  = threadIdx.x;
  const int  lane = tid & 31;
  const int  wave = tid >> 5;
  const int  half = lane >> 4;
  const int  lrow = lane & 15;
  const long rowBase = (long)blockIdx.x * 128;

  v8f acc[8] = {};

  const int ksteps = (K + 31) >> 5;
  for (int ks = 0; ks < ksteps; ++ks) {
    const int k0 = ks << 5;
    __syncthreads();

    // ---- stage A tile (128 rows x 32 k): float4 global load -> v4bf LDS store ----
#pragma unroll
    for (int t = 0; t < 4; ++t) {
      int  qi = t * 256 + tid;        // quad index, 1024 quads
      int  rl = qi >> 3;              // row in tile (8 quads per row)
      int  kl = (qi & 7) << 2;        // k within tile, multiple of 4
      int  kg = k0 + kl;
      long rg = rowBase + rl;
      float4 v = make_float4(0.f, 0.f, 0.f, 0.f);
      if (kg < K) {
        const float* src = nullptr;
        if (MODE == MODE_PLAIN) {
          src = A + rg * (long)K + kg;
        } else {
          int e = (int)(rg % NEDGE);
          int b = (int)(rg / NEDGE);
          if (MODE == MODE_N2E) {
            int node = (kg < HD) ? ridx[e] : sidx[e];
            int kk   = (kg < HD) ? kg : (kg - HD);
            src = nodeSrc + ((long)b * NNODE + node) * HD + kk;
          } else { // MODE_OUT
            if (kg < 2 * HD) {
              int node = (kg < HD) ? ridx[e] : sidx[e];
              int kk   = (kg < HD) ? kg : (kg - HD);
              src = nodeSrc + ((long)b * NNODE + node) * HD + kk;
            } else {
              src = A + rg * (long)HD + (kg - 2 * HD);
            }
          }
        }
        v = *(const float4*)src;
      }
      int hf = (kl >> 3) & 1;
      int ib = ((kl >> 4) << 3) + (kl & 7);     // fragment elem base (i..i+3)
      int ln = (hf << 4) + (rl & 15);
      int wv = rl >> 4;
      v4bf q; q[0] = f2bf(v.x); q[1] = f2bf(v.y); q[2] = f2bf(v.z); q[3] = f2bf(v.w);
      *(v4bf*)&lAe[(((wv << 5) + ln) << 4) + ib] = q;
    }

    // ---- stage B tile (32 k x 128 n): 4 strided loads -> v4bf LDS store ----
#pragma unroll
    for (int t = 0; t < 4; ++t) {
      int qi = t * 256 + tid;          // 1024 quads
      int n  = qi & 127;
      int kl = (qi >> 7) << 2;         // k within tile, multiple of 4
      int kg = k0 + kl;
      float4 v = make_float4(0.f, 0.f, 0.f, 0.f);
      if (kg < K) {
        const float* src = W + (long)kg * HD + n;
        v.x = src[0 * HD]; v.y = src[1 * HD]; v.z = src[2 * HD]; v.w = src[3 * HD];
      }
      int hf = (kl >> 3) & 1;
      int ib = ((kl >> 4) << 3) + (kl & 7);
      int ln = (hf << 4) + (n & 15);
      int nt = n >> 4;
      v4bf q; q[0] = f2bf(v.x); q[1] = f2bf(v.y); q[2] = f2bf(v.z); q[3] = f2bf(v.w);
      *(v4bf*)&lBe[(((nt << 5) + ln) << 4) + ib] = q;
    }
    __syncthreads();

    // ---- WMMA: one contiguous 32B LDS load per fragment ----
    v16bf af = lAfrag[(wave << 5) + lane];
#pragma unroll
    for (int nt = 0; nt < 8; ++nt) {
      v16bf bfr = lBfrag[(nt << 5) + lane];
      acc[nt] = __builtin_amdgcn_wmma_f32_16x16x32_bf16(
          false, af, false, bfr, (short)0, acc[nt], false, false);
    }
  }

  // ---- epilogue: bias + ELU, f32 store (C/D layout: VGPR r -> M=r / r+8) ----
#pragma unroll
  for (int nt = 0; nt < 8; ++nt) {
    int col = nt * 16 + lrow;
    float bv = bias[col];
#pragma unroll
    for (int r = 0; r < 8; ++r) {
      long row = rowBase + wave * 16 + half * 8 + r;
      float v = acc[nt][r] + bv;
      v = (v > 0.0f) ? v : (__expf(v) - 1.0f);
      out[row * HD + col] = v;
    }
  }
}

__global__ void build_idx_kernel(const float* __restrict__ rr, const float* __restrict__ rs,
                                 int* __restrict__ ridx, int* __restrict__ sidx) {
  int e = blockIdx.x * blockDim.x + threadIdx.x;
  if (e >= NEDGE) return;
  int r = 0, s = 0;
  for (int n = 0; n < NNODE; ++n) {
    if (rr[e * NNODE + n] > 0.5f) r = n;
    if (rs[e * NNODE + n] > 0.5f) s = n;
  }
  ridx[e] = r;
  sidx[e] = s;
}

__global__ void zero_kernel(float* __restrict__ p, long n) {
  long i = (long)blockIdx.x * blockDim.x + threadIdx.x;
  if (i < n) p[i] = 0.0f;
}

__global__ __launch_bounds__(128) void bn_stats_kernel(const float* __restrict__ x,
                                                       float* __restrict__ sums, int M) {
  int col = threadIdx.x;
  float s = 0.f, s2 = 0.f;
  for (long r = blockIdx.x; r < M; r += gridDim.x) {
    float v = x[r * HD + col];
    s += v; s2 += v * v;
  }
  atomicAdd(&sums[col], s);
  atomicAdd(&sums[HD + col], s2);
}

__global__ __launch_bounds__(128) void bn_apply_kernel(float* __restrict__ x,
                                                       const float* __restrict__ sums,
                                                       const float* __restrict__ g,
                                                       const float* __restrict__ bt, int M) {
  int col = threadIdx.x;
  float m  = sums[col] / (float)M;
  float vv = sums[HD + col] / (float)M - m * m;     // biased variance
  float sc = rsqrtf(vv + 1e-5f) * g[col];
  float sh = bt[col] - m * sc;
  for (long r = blockIdx.x; r < M; r += gridDim.x)
    x[r * HD + col] = x[r * HD + col] * sc + sh;
}

__global__ void edge2node_kernel(const float* __restrict__ xe, float* __restrict__ xn,
                                 const int* __restrict__ ridx) {
  long i = (long)blockIdx.x * blockDim.x + threadIdx.x;
  const long total = (long)NBATCH * NEDGE * HD;
  if (i >= total) return;
  int  h  = (int)(i & (HD - 1));
  long re = i >> 7;
  int  e  = (int)(re % NEDGE);
  int  b  = (int)(re / NEDGE);
  atomicAdd(&xn[((long)b * NNODE + ridx[e]) * HD + h], xe[i] * (1.0f / NNODE));
}

__global__ void fc_kernel(const float* __restrict__ x, const float* __restrict__ w,
                          const float* __restrict__ b, float* __restrict__ out) {
  long i = (long)blockIdx.x * blockDim.x + threadIdx.x;
  const long total = (long)NBATCH * NEDGE * 4;
  if (i >= total) return;
  int  j = (int)(i & 3);
  long r = i >> 2;
  float acc = b[j];
  const float* xr = x + r * HD;
#pragma unroll 8
  for (int k = 0; k < HD; ++k) acc += xr[k] * w[k * 4 + j];
  out[i] = acc;
}

extern "C" void kernel_launch(void* const* d_in, const int* in_sizes, int n_in,
                              void* d_out, int out_size, void* d_ws, size_t ws_size,
                              hipStream_t stream) {
  (void)in_sizes; (void)n_in; (void)out_size; (void)ws_size;
  const float* inputs   = (const float*)d_in[0];
  const float* rel_rec  = (const float*)d_in[1];
  const float* rel_send = (const float*)d_in[2];
  const float *emb_w1 = (const float*)d_in[3],  *emb_b1 = (const float*)d_in[4],
              *emb_w2 = (const float*)d_in[5],  *emb_b2 = (const float*)d_in[6],
              *emb_g  = (const float*)d_in[7],  *emb_bt = (const float*)d_in[8];
  const float *n2e_w1 = (const float*)d_in[9],  *n2e_b1 = (const float*)d_in[10],
              *n2e_w2 = (const float*)d_in[11], *n2e_b2 = (const float*)d_in[12],
              *n2e_g  = (const float*)d_in[13], *n2e_bt = (const float*)d_in[14];
  const float *e2n_w1 = (const float*)d_in[15], *e2n_b1 = (const float*)d_in[16],
              *e2n_w2 = (const float*)d_in[17], *e2n_b2 = (const float*)d_in[18],
              *e2n_g  = (const float*)d_in[19], *e2n_bt = (const float*)d_in[20];
  const float *out_w1 = (const float*)d_in[21], *out_b1 = (const float*)d_in[22],
              *out_w2 = (const float*)d_in[23], *out_b2 = (const float*)d_in[24],
              *out_g  = (const float*)d_in[25], *out_bt = (const float*)d_in[26];
  const float *fc_w = (const float*)d_in[27], *fc_b = (const float*)d_in[28];

  const int MN = NBATCH * NNODE;            // 4096 node rows
  const int ME = NBATCH * NEDGE;            // 258048 edge rows

  char* w = (char*)d_ws;
  auto alloc = [&](size_t bytes) -> char* {
    char* p = w; w += (bytes + 255) & ~(size_t)255; return p;
  };
  int*   ridx = (int*)  alloc((size_t)NEDGE * 4);
  int*   sidx = (int*)  alloc((size_t)NEDGE * 4);
  float* sums = (float*)alloc(256 * 4);
  float* nA   = (float*)alloc((size_t)MN * HD * 4);   // node activations (BN out)
  float* nT   = (float*)alloc((size_t)MN * HD * 4);   // node hidden temp
  float* nB   = (float*)alloc((size_t)MN * HD * 4);   // edge2node scatter target
  float* eH   = (float*)alloc((size_t)ME * HD * 4);   // edge hidden (132 MB)
  float* eS   = (float*)alloc((size_t)ME * HD * 4);   // edge BN out / skip (132 MB)

  build_idx_kernel<<<(NEDGE + 255) / 256, 256, 0, stream>>>(rel_rec, rel_send, ridx, sidx);

  // ----- emb MLP: [4096,196] -> [4096,128] -----
  gemm_bias_elu<MODE_PLAIN><<<MN / 128, 256, 0, stream>>>(inputs, nullptr, emb_w1, emb_b1, ridx, sidx, nT, MN, 196);
  gemm_bias_elu<MODE_PLAIN><<<MN / 128, 256, 0, stream>>>(nT, nullptr, emb_w2, emb_b2, ridx, sidx, nA, MN, HD);
  zero_kernel<<<1, 256, 0, stream>>>(sums, 256);
  bn_stats_kernel<<<64, 128, 0, stream>>>(nA, sums, MN);
  bn_apply_kernel<<<64, 128, 0, stream>>>(nA, sums, emb_g, emb_bt, MN);

  // ----- node2edge (fused) + n2e MLP: [258048,256] -> [258048,128] -----
  gemm_bias_elu<MODE_N2E><<<ME / 128, 256, 0, stream>>>(nullptr, nA, n2e_w1, n2e_b1, ridx, sidx, eH, ME, 2 * HD);
  gemm_bias_elu<MODE_PLAIN><<<ME / 128, 256, 0, stream>>>(eH, nullptr, n2e_w2, n2e_b2, ridx, sidx, eS, ME, HD);
  zero_kernel<<<1, 256, 0, stream>>>(sums, 256);
  bn_stats_kernel<<<1024, 128, 0, stream>>>(eS, sums, ME);
  bn_apply_kernel<<<1024, 128, 0, stream>>>(eS, sums, n2e_g, n2e_bt, ME);   // eS == x_skip

  // ----- edge2node scatter (/N) -----
  zero_kernel<<<((long)MN * HD + 255) / 256, 256, 0, stream>>>(nB, (long)MN * HD);
  edge2node_kernel<<<((long)ME * HD + 255) / 256, 256, 0, stream>>>(eS, nB, ridx);

  // ----- e2n MLP: [4096,128] -> [4096,128] -----
  gemm_bias_elu<MODE_PLAIN><<<MN / 128, 256, 0, stream>>>(nB, nullptr, e2n_w1, e2n_b1, ridx, sidx, nT, MN, HD);
  gemm_bias_elu<MODE_PLAIN><<<MN / 128, 256, 0, stream>>>(nT, nullptr, e2n_w2, e2n_b2, ridx, sidx, nA, MN, HD);
  zero_kernel<<<1, 256, 0, stream>>>(sums, 256);
  bn_stats_kernel<<<64, 128, 0, stream>>>(nA, sums, MN);
  bn_apply_kernel<<<64, 128, 0, stream>>>(nA, sums, e2n_g, e2n_bt, MN);

  // ----- node2edge + skip concat (fused) + out MLP: [258048,384] -> [258048,128] -----
  gemm_bias_elu<MODE_OUT><<<ME / 128, 256, 0, stream>>>(eS, nA, out_w1, out_b1, ridx, sidx, eH, ME, 3 * HD);
  gemm_bias_elu<MODE_PLAIN><<<ME / 128, 256, 0, stream>>>(eH, nullptr, out_w2, out_b2, ridx, sidx, eS, ME, HD);
  zero_kernel<<<1, 256, 0, stream>>>(sums, 256);
  bn_stats_kernel<<<1024, 128, 0, stream>>>(eS, sums, ME);
  bn_apply_kernel<<<1024, 128, 0, stream>>>(eS, sums, out_g, out_bt, ME);

  // ----- final fc: [258048,128] @ [128,4] + b -----
  fc_kernel<<<((long)ME * 4 + 255) / 256, 256, 0, stream>>>(eS, fc_w, fc_b, (float*)d_out);
}
